// KSG_critic_3736621548242
// MI455X (gfx1250) — compile-verified, matrix operand order checked by reading, not run on hardware
//
#include <hip/hip_runtime.h>
#include <stdint.h>

// ---------------------------------------------------------------------------
// KSG mutual-information estimator for N=4096, dx=dy=128 (float32 in/out).
// VALU-bound Chebyshev distances with TDM (tensor_load_to_lds) double-buffered
// LDS tiling; radius counts summed with V_WMMA_I32_16X16X64_IU8.
// 4 waves per workgroup share B tiles; wave 0 drives the TDM.
// ---------------------------------------------------------------------------

typedef __attribute__((ext_vector_type(4))) float    f32x4;
typedef __attribute__((ext_vector_type(4))) int      v4i;
typedef __attribute__((ext_vector_type(8))) int      v8i;
typedef __attribute__((ext_vector_type(4))) unsigned v4u;

#ifndef __has_builtin
#define __has_builtin(x) 0
#endif

#if __has_builtin(__builtin_amdgcn_tensor_load_to_lds) && \
    __has_builtin(__builtin_amdgcn_s_wait_tensorcnt)
#define USE_TDM 1
#else
#define USE_TDM 0
#endif

__device__ __forceinline__ void sched_bar() {
#if __has_builtin(__builtin_amdgcn_sched_barrier)
  __builtin_amdgcn_sched_barrier(0);   // cap load-clustering / VGPR pressure
#else
  asm volatile("" ::: "memory");
#endif
}

constexpr int NPTS  = 4096;
constexpr int DIM   = 128;
constexpr int TI    = 16;        // rows per wave
constexpr int WVS   = 4;         // waves per workgroup
constexpr int ROWS_WG = TI * WVS;        // 64 rows per workgroup
constexpr int AST   = DIM + 4;   // padded LDS row stride for A tiles (132 dwords)
constexpr int TJ1   = 32;        // knn col-tile
constexpr int TJ2   = 64;        // count col-tile (= WMMA K)
constexpr int NSEG1 = 4;         // j-segments for knn pass
constexpr int NSEG2 = 8;         // j-segments for count pass

// ----------------------------- TDM tile loader ------------------------------
// Loads `rows` x DIM float tile (row-major, global row stride = DIM elements)
// into LDS. padA=true inserts 4 dwords of LDS padding per 128-dword row
// (TDM pad_interval=6 -> every 512B, pad_amount=3 -> 4 dwords) giving LDS
// stride AST=132 for bank-conflict-free per-lane row access.
__device__ __forceinline__ void tile_load(float* lds_dst, const float* gsrc,
                                          int rows, bool padA) {
#if USE_TDM
  const uint64_t ga  = (uint64_t)(uintptr_t)gsrc;
  const unsigned lds = (unsigned)(uintptr_t)lds_dst; // low 32b = LDS byte offset
  v4u g0;
  g0[0] = 1u;                                                  // count=1 (user D#)
  g0[1] = lds;                                                 // lds_addr
  g0[2] = (unsigned)ga;                                        // global_addr[31:0]
  g0[3] = (unsigned)((ga >> 32) & 0x01FFFFFFu) | (2u << 30);   // [56:32] | type=2

  const unsigned td0 = (unsigned)DIM;     // tensor_dim0 (elements per row)
  const unsigned td1 = (unsigned)rows;    // tensor_dim1
  const unsigned pad = padA ? ((1u << 20) | (6u << 22) | (3u << 25)) : 0u;
  v8i g1;
  g1[0] = (int)((2u << 16) | pad);                       // data_size=4B (+pad cfg)
  g1[1] = (int)((td0 & 0xFFFFu) << 16);                  // tensor_dim0[15:0]
  g1[2] = (int)((td0 >> 16) | ((td1 & 0xFFFFu) << 16));  // dim0 hi | dim1 lo
  g1[3] = (int)((td1 >> 16) | ((unsigned)DIM << 16));    // dim1 hi | tile_dim0
  g1[4] = (int)((unsigned)rows & 0xFFFFu);               // tile_dim1, tile_dim2=0
  g1[5] = (int)(unsigned)DIM;                            // tensor_dim0_stride
  g1[6] = 0;                                             // stride hi | dim1_stride
  g1[7] = 0;
  const v4i z4 = {0, 0, 0, 0};
#if defined(__clang_major__) && (__clang_major__ >= 23)
  const v8i z8 = {0, 0, 0, 0, 0, 0, 0, 0};
  __builtin_amdgcn_tensor_load_to_lds(g0, g1, z4, z4, z8, 0);
#else
  __builtin_amdgcn_tensor_load_to_lds(g0, g1, z4, z4, 0);
#endif
#else
  const int stride = padA ? AST : DIM;
  for (int idx = threadIdx.x & 31; idx < rows * DIM; idx += 32) {
    const int rr = idx >> 7, cc = idx & 127;
    lds_dst[rr * stride + cc] = gsrc[(size_t)rr * DIM + cc];
  }
#endif
}

// Wait for this wave's outstanding TDM ops (no-op for non-issuing waves).
__device__ __forceinline__ void tdm_wait() {
#if USE_TDM
  __builtin_amdgcn_s_wait_tensorcnt(0);
  asm volatile("" ::: "memory");
#endif
}

// ---------------------------- top-5 maintenance -----------------------------
__device__ __forceinline__ void top5_insert(float (&t)[5], float v) {
  if (v > t[4]) {                 // rarely taken: amortized over 512-deep dists
    t[4] = v;
#pragma unroll
    for (int q = 4; q > 0; --q) {
      if (t[q] > t[q - 1]) { float tmp = t[q - 1]; t[q - 1] = t[q]; t[q] = tmp; }
    }
  }
}

// ---------------------------------------------------------------------------
// Kernel 1: per-row top-5 LARGEST joint Chebyshev distance, partial over a
// j-segment.  4 waves/wg, wave w owns rows w*16..w*16+15; B tiles shared.
// Lane L -> row (L&15), col half (L>>4).
// ---------------------------------------------------------------------------
__global__ __launch_bounds__(WVS * 32)
__attribute__((amdgpu_waves_per_eu(4)))
void knn_partial_kernel(const float* __restrict__ x,
                        const float* __restrict__ y,
                        float* __restrict__ top5_out) {
  __shared__ __align__(16) float ax[ROWS_WG * AST];
  __shared__ __align__(16) float ay[ROWS_WG * AST];
  __shared__ __align__(16) float bx[2][TJ1 * DIM];
  __shared__ __align__(16) float by[2][TJ1 * DIM];
  __shared__ float red[WVS * 32 * 5];

  const int tid  = threadIdx.x;
  const int wave = tid >> 5;
  const int lane = tid & 31;
  const int m    = lane & 15;
  const int hi   = lane >> 4;
  const int seg  = blockIdx.y;
  const int wg_row0 = blockIdx.x * ROWS_WG;
  const int row     = wg_row0 + wave * TI + m;   // this lane's row
  const int j0      = seg * (NPTS / NSEG1);
  const int ntiles  = (NPTS / NSEG1) / TJ1;

  if (wave == 0) {
    tile_load(&ax[0], x + (size_t)wg_row0 * DIM, ROWS_WG, true);
    tile_load(&ay[0], y + (size_t)wg_row0 * DIM, ROWS_WG, true);
    tile_load(&bx[0][0], x + (size_t)j0 * DIM, TJ1, false);
    tile_load(&by[0][0], y + (size_t)j0 * DIM, TJ1, false);
  }
  tdm_wait();
  __syncthreads();

  float t5[5] = {-1.f, -1.f, -1.f, -1.f, -1.f};
  float acc[16];
#pragma unroll
  for (int c = 0; c < 16; ++c) acc[c] = 0.f;

  const int arow = (wave * TI + m) * AST;

  for (int t = 0; t < ntiles; ++t) {
    const int buf = t & 1;
    if (wave == 0 && t + 1 < ntiles) {   // async prefetch while computing
      const int nb = buf ^ 1;
      const size_t jn = (size_t)(j0 + (t + 1) * TJ1) * DIM;
      tile_load(&bx[nb][0], x + jn, TJ1, false);
      tile_load(&by[nb][0], y + jn, TJ1, false);
    }
#pragma unroll 1
    for (int part = 0; part < 2; ++part) {  // joint dist = max over x then y dims
      const float* a = (part == 0) ? &ax[arow] : &ay[arow];
      const float* b = (part == 0) ? &bx[buf][hi * 16 * DIM]
                                   : &by[buf][hi * 16 * DIM];
#pragma unroll 1
      for (int dc = 0; dc < DIM / 8; ++dc) {
        const f32x4 a0 = *(const f32x4*)(a + dc * 8);
        const f32x4 a1 = *(const f32x4*)(a + dc * 8 + 4);
#pragma unroll
        for (int c = 0; c < 16; ++c) {
          const float* bb = b + c * DIM + dc * 8;   // lane-broadcast reads
          const f32x4 b0 = *(const f32x4*)(bb);
          const f32x4 b1 = *(const f32x4*)(bb + 4);
          float v = acc[c];
#pragma unroll
          for (int e = 0; e < 4; ++e) {
            v = fmaxf(v, fabsf(a0[e] - b0[e]));
            v = fmaxf(v, fabsf(a1[e] - b1[e]));
          }
          acc[c] = v;
          if ((c & 3) == 3) sched_bar();   // bound in-flight ds_load_b128s
        }
      }
    }
#pragma unroll
    for (int c = 0; c < 16; ++c) {
      top5_insert(t5, acc[c]);
      acc[c] = 0.f;
    }
    tdm_wait();      // prefetched tiles have landed (wave 0)
    __syncthreads(); // publish to all waves; all done reading old buffer
  }

  // merge lane pair (L, L+16) -> partial top-5 for this row
#pragma unroll
  for (int q = 0; q < 5; ++q) red[tid * 5 + q] = t5[q];
  __syncthreads();
  if (hi == 0) {
    const float* A5 = &red[(wave * 32 + m) * 5];
    const float* B5 = &red[(wave * 32 + m + 16) * 5];
    float outv[5];
    int i = 0, j = 0;
#pragma unroll
    for (int q = 0; q < 5; ++q) {
      const float av = (i < 5) ? A5[i] : -3.0e38f;
      const float bv = (j < 5) ? B5[j] : -3.0e38f;
      if (av >= bv) { outv[q] = av; ++i; } else { outv[q] = bv; ++j; }
    }
    float* dst = top5_out + ((size_t)seg * NPTS + row) * 5;
#pragma unroll
    for (int q = 0; q < 5; ++q) dst[q] = outv[q];
  }
}

// ---------------------------------------------------------------------------
// Kernel 2: merge per-segment top-5 -> knn radius r and log r per row.
// ---------------------------------------------------------------------------
__global__ void knn_merge_kernel(const float* __restrict__ top5,
                                 float* __restrict__ r,
                                 float* __restrict__ logr) {
  const int row = blockIdx.x * blockDim.x + threadIdx.x;
  if (row >= NPTS) return;
  float t[5] = {-1.f, -1.f, -1.f, -1.f, -1.f};
  for (int s = 0; s < NSEG1; ++s) {
    const float* p = top5 + ((size_t)s * NPTS + row) * 5;
#pragma unroll
    for (int q = 0; q < 5; ++q) top5_insert(t, p[q]);
  }
  const float knn = t[4];            // 5th largest (torch.topk largest=True)
  r[row]    = knn + 1e-15f;
  logr[row] = logf(knn);
}

// ---------------------------------------------------------------------------
// Kernel 3: marginal radius counts.  Indicator bytes are built directly in the
// V_WMMA_I32_16X16X64_IU8 A-matrix layout; B = all-ones bytes => D rows hold
// per-row counts.  4 waves/wg share B tiles; counts merged atomically.
// ---------------------------------------------------------------------------
__global__ __launch_bounds__(WVS * 32)
__attribute__((amdgpu_waves_per_eu(4)))
void count_kernel(const float* __restrict__ data, const float* __restrict__ r,
                  int* __restrict__ cnt) {
  __shared__ __align__(16) float asm_[ROWS_WG * AST];
  __shared__ __align__(16) float bsm[2][TJ2 * DIM];

  const int tid  = threadIdx.x;
  const int wave = tid >> 5;
  const int lane = tid & 31;
  const int m    = lane & 15;
  const int hi   = lane >> 4;
  const int seg  = blockIdx.y;
  const int wg_row0 = blockIdx.x * ROWS_WG;
  const int row0    = wg_row0 + wave * TI;       // this wave's first row
  const int j0      = seg * (NPTS / NSEG2);
  const int ntiles  = (NPTS / NSEG2) / TJ2;

  const float rr = r[row0 + m];

  if (wave == 0) {
    tile_load(&asm_[0], data + (size_t)wg_row0 * DIM, ROWS_WG, true);
    tile_load(&bsm[0][0], data + (size_t)j0 * DIM, TJ2, false);
  }
  tdm_wait();
  __syncthreads();

  v8i dacc = {0, 0, 0, 0, 0, 0, 0, 0};
  const v8i bones = {0x01010101, 0x01010101, 0x01010101, 0x01010101,
                     0x01010101, 0x01010101, 0x01010101, 0x01010101};

  float acc[32];
#pragma unroll
  for (int s = 0; s < 32; ++s) acc[s] = 0.f;

  const int arow = (wave * TI + m) * AST;

  for (int t = 0; t < ntiles; ++t) {
    const int buf = t & 1;
    if (wave == 0 && t + 1 < ntiles)
      tile_load(&bsm[buf ^ 1][0],
                data + (size_t)(j0 + (t + 1) * TJ2) * DIM, TJ2, false);

    const float* a = &asm_[arow];
    const float* b = &bsm[buf][0];
#pragma unroll 1
    for (int dc = 0; dc < DIM / 8; ++dc) {
      const f32x4 a0 = *(const f32x4*)(a + dc * 8);
      const f32x4 a1 = *(const f32x4*)(a + dc * 8 + 4);
#pragma unroll
      for (int v = 0; v < 8; ++v) {
        // IU8 A-matrix K for (lane, vgpr v, byte): K = g*16 + hi*8 + p*4 + byte
        const int cb = (v >> 1) * 16 + hi * 8 + (v & 1) * 4;
#pragma unroll
        for (int byt = 0; byt < 4; ++byt) {
          const float* bb = b + (cb + byt) * DIM + dc * 8;  // 2-addr broadcast
          const f32x4 b0 = *(const f32x4*)(bb);
          const f32x4 b1 = *(const f32x4*)(bb + 4);
          float vv = acc[v * 4 + byt];
#pragma unroll
          for (int e = 0; e < 4; ++e) {
            vv = fmaxf(vv, fabsf(a0[e] - b0[e]));
            vv = fmaxf(vv, fabsf(a1[e] - b1[e]));
          }
          acc[v * 4 + byt] = vv;
        }
        sched_bar();   // bound in-flight ds_load_b128s (8 per group)
      }
    }

    // pack indicators (dist <= r) into IU8 A layout and accumulate with WMMA
    v8i amat;
#pragma unroll
    for (int v = 0; v < 8; ++v) {
      unsigned p = 0;
#pragma unroll
      for (int byt = 0; byt < 4; ++byt) {
        p |= (acc[v * 4 + byt] <= rr ? 1u : 0u) << (8 * byt);
        acc[v * 4 + byt] = 0.f;
      }
      amat[v] = (int)p;
    }
    dacc = __builtin_amdgcn_wmma_i32_16x16x64_iu8(false, amat, false, bones,
                                                  dacc, false, false);
    tdm_wait();
    __syncthreads();
  }

  // D layout: lanes 0-15 vgpr v -> row v; lanes 16-31 -> row v+8 (all cols equal)
  if (m == 0) {
    const int base = row0 + hi * 8;
#pragma unroll
    for (int v = 0; v < 8; ++v) atomicAdd(&cnt[base + v], dacc[v]);
  }
}

// ---------------------------------------------------------------------------
// Kernel 4: final reduction + KSG combine (k=5 per reference setup).
// ---------------------------------------------------------------------------
__global__ void final_kernel(const float* __restrict__ logr,
                             const int* __restrict__ cntx,
                             const int* __restrict__ cnty,
                             float* __restrict__ out) {
  __shared__ float s0[256], s1[256], s2[256];
  const int tid = threadIdx.x;
  float a = 0.f, b = 0.f, c = 0.f;
  for (int i = tid; i < NPTS; i += 256) {
    a += logr[i];
    b += logf((float)cntx[i] - 1.0f);
    c += logf((float)cnty[i] - 1.0f);
  }
  s0[tid] = a; s1[tid] = b; s2[tid] = c;
  __syncthreads();
  for (int off = 128; off > 0; off >>= 1) {
    if (tid < off) {
      s0[tid] += s0[tid + off];
      s1[tid] += s1[tid + off];
      s2[tid] += s2[tid + off];
    }
    __syncthreads();
  }
  if (tid == 0) {
    const float invN  = 1.0f / (float)NPTS;
    const float mlr   = s0[0] * invN;
    const float mlx   = s1[0] * invN;
    const float mly   = s2[0] * invN;
    const float LOG2C = 0.69314718055994530942f;
    const float LOGN  = 8.3177661667193433f;      // log(4096)
    const float DIG5  = 1.5061176684318003f;      // digamma(5)
    const float ans_xy = -DIG5 + LOGN + 256.f * LOG2C + 256.f * mlr;
    const float ans_x  =  LOGN + 128.f * LOG2C - mlx + 128.f * mlr;
    const float ans_y  =  LOGN + 128.f * LOG2C - mly + 128.f * mlr;
    out[0] = ans_x + ans_y - ans_xy;
  }
}

// ---------------------------------------------------------------------------
extern "C" void kernel_launch(void* const* d_in, const int* in_sizes, int n_in,
                              void* d_out, int out_size, void* d_ws,
                              size_t ws_size, hipStream_t stream) {
  (void)in_sizes; (void)n_in; (void)out_size; (void)ws_size;
  const float* x = (const float*)d_in[0];
  const float* y = (const float*)d_in[1];
  float* out = (float*)d_out;

  char* w = (char*)d_ws;
  const size_t top5_bytes = (size_t)NSEG1 * NPTS * 5 * sizeof(float); // 320 KB
  float* top5 = (float*)w;
  float* r    = (float*)(w + top5_bytes);
  float* logr = r + NPTS;
  int*   cntx = (int*)(logr + NPTS);
  int*   cnty = cntx + NPTS;

  hipMemsetAsync(cntx, 0, 2 * (size_t)NPTS * sizeof(int), stream);

  dim3 g1(NPTS / ROWS_WG, NSEG1);
  knn_partial_kernel<<<g1, WVS * 32, 0, stream>>>(x, y, top5);
  knn_merge_kernel<<<NPTS / 256, 256, 0, stream>>>(top5, r, logr);

  dim3 g2(NPTS / ROWS_WG, NSEG2);
  count_kernel<<<g2, WVS * 32, 0, stream>>>(x, r, cntx);
  count_kernel<<<g2, WVS * 32, 0, stream>>>(y, r, cnty);

  final_kernel<<<1, 256, 0, stream>>>(logr, cntx, cnty, out);
}
